// MMFSBlock_74801150427597
// MI455X (gfx1250) — matrix operand order repaired and checked by reading, not compiled, not run on hardware
//
#include <hip/hip_runtime.h>

typedef __attribute__((ext_vector_type(16))) _Float16 v16h;
typedef __attribute__((ext_vector_type(8)))  _Float16 v8h;
typedef __attribute__((ext_vector_type(8)))  float    v8f;

namespace {
constexpr int kB    = 2;
constexpr int kQD   = 320;    // query channels
constexpr int kFD   = 1024;   // feature dim
constexpr int kDM   = 1024;   // value dim (NH*HD)
constexpr int kNH   = 16;
constexpr int kHD   = 64;
constexpr int kNP   = 8;
constexpr int kN    = 8;      // images
constexpr int kMAXN = 10;
constexpr int kLq   = 1024;   // 32x32 queries
constexpr int kHv   = 16;     // value spatial
constexpr int kTok  = 256;    // kHv*kHv
constexpr int kOffC = kNH * kNP * 2;            // 256
constexpr int kLogC = kNH * kMAXN * kNP;        // 1280
}

// Async global->LDS copy of 16 bytes (CDNA5, tracked by ASYNCcnt).
// VDST = LDS byte address (low 32 bits of flat shared address), VADDR = 64b.
__device__ __forceinline__ void async_copy_b128(unsigned ldsOff,
                                                const _Float16* gp) {
  asm volatile("global_load_async_to_lds_b128 %0, %1, off"
               :: "v"(ldsOff), "v"(gp)
               : "memory");
}
__device__ __forceinline__ void wait_async0() {
  asm volatile("s_wait_asynccnt 0x0" ::: "memory");
}

// ---------------- query layernorm + sincos positional embedding -> f16 ------
__global__ __launch_bounds__(kQD)
void prep_query_kernel(const float* __restrict__ sample,
                       const float* __restrict__ lnw,
                       const float* __restrict__ lnb,
                       _Float16* __restrict__ qH)
{
  __shared__ float sh1[10], sh2[10], mom[2];
  const int bq = blockIdx.x;            // b*1024 + q
  const int b  = bq >> 10;
  const int q  = bq & 1023;
  const int c  = threadIdx.x;           // channel 0..319
  const float x = sample[((size_t)b * kQD + c) * (size_t)kLq + q];
  float s = x, s2 = x * x;
#pragma unroll
  for (int off = 16; off > 0; off >>= 1) {
    s  += __shfl_xor(s,  off, 32);
    s2 += __shfl_xor(s2, off, 32);
  }
  const int wid = c >> 5, lane = c & 31;
  if (lane == 0) { sh1[wid] = s; sh2[wid] = s2; }
  __syncthreads();
  if (c == 0) {
    float t1 = 0.f, t2 = 0.f;
    for (int i = 0; i < 10; ++i) { t1 += sh1[i]; t2 += sh2[i]; }
    mom[0] = t1 * (1.f / kQD);
    mom[1] = t2 * (1.f / kQD);
  }
  __syncthreads();
  const float mean = mom[0];
  const float rs   = rsqrtf(mom[1] - mean * mean + 1e-6f);
  float v = (x - mean) * rs * lnw[c] + lnb[c];
  // sincos pos embed: c<160 uses row index, else col index; within each 160,
  // first 80 are sin, next 80 cos, omega_k = 10000^(-k/80)
  const int gy = q >> 5, gx = q & 31;
  const float pos = (c < 160) ? (float)gy : (float)gx;
  int cc = (c < 160) ? c : (c - 160);
  const bool is_cos = cc >= 80;
  if (is_cos) cc -= 80;
  const float omega = __powf(10000.f, -(float)cc * (1.f / 80.f));
  const float ang = pos * omega;
  v += is_cos ? __cosf(ang) : __sinf(ang);
  qH[(size_t)bq * kQD + c] = (_Float16)v;
}

// ---------------- feature layernorm -> f16 ----------------------------------
__global__ __launch_bounds__(256)
void prep_feat_kernel(const float* __restrict__ feat,
                      const float* __restrict__ lnw,
                      const float* __restrict__ lnb,
                      _Float16* __restrict__ featH)
{
  __shared__ float sh1[8], sh2[8], mom[2];
  const size_t row = blockIdx.x;        // b*N*256 rows
  const float* src = feat + row * kFD;
  float vals[4];
  float s = 0.f, s2 = 0.f;
#pragma unroll
  for (int j = 0; j < 4; ++j) {
    float x = src[threadIdx.x + j * 256];
    vals[j] = x; s += x; s2 += x * x;
  }
#pragma unroll
  for (int off = 16; off > 0; off >>= 1) {
    s  += __shfl_xor(s,  off, 32);
    s2 += __shfl_xor(s2, off, 32);
  }
  const int wid = threadIdx.x >> 5, lane = threadIdx.x & 31;
  if (lane == 0) { sh1[wid] = s; sh2[wid] = s2; }
  __syncthreads();
  if (threadIdx.x == 0) {
    float t1 = 0.f, t2 = 0.f;
    for (int i = 0; i < 8; ++i) { t1 += sh1[i]; t2 += sh2[i]; }
    mom[0] = t1 * (1.f / kFD);
    mom[1] = t2 * (1.f / kFD);
  }
  __syncthreads();
  const float mean = mom[0];
  const float rs   = rsqrtf(mom[1] - mean * mean + 1e-6f);
  _Float16* dst = featH + row * kFD;
#pragma unroll
  for (int j = 0; j < 4; ++j) {
    int c = threadIdx.x + j * 256;
    dst[c] = (_Float16)((vals[j] - mean) * rs * lnw[c] + lnb[c]);
  }
}

// ---------------- f32 -> f16 weight converts --------------------------------
__global__ void cvt_f16_kernel(const float* __restrict__ src,
                               _Float16* __restrict__ dst, int n)
{
  int i = blockIdx.x * 256 + threadIdx.x;
  if (i < n) dst[i] = (_Float16)src[i];
}

__global__ void cvt_f16_transpose_kernel(const float* __restrict__ src,
                                         _Float16* __restrict__ dst,
                                         int rows, int cols)
{
  // src: rows x cols, dst: cols x rows (dst[c*rows+r] = src[r*cols+c])
  int i = blockIdx.x * 256 + threadIdx.x;
  if (i < rows * cols) {
    int c = i / rows, r = i % rows;
    dst[(size_t)c * rows + r] = (_Float16)src[(size_t)r * cols + c];
  }
}

// ---------------- WMMA GEMM: C(MxN,f32|f16) = A(MxK,f16) * B(KxN,f16) + bias
// Block = 4 waves = 128x64 output tile; wave computes 32x64 (2 A frags x 4 B
// tiles = 8 WMMAs per K-step). The shared 32x64 B tile is staged into LDS by
// async global->LDS copies (double buffered): s_wait_asynccnt + barrier,
// issue next tile, compute current from LDS. All dims are exact multiples, so EXEC
// is all-ones at every WMMA.
template<bool OUT_F16>
__global__ __launch_bounds__(128)
void gemm_wmma(const _Float16* __restrict__ A, const _Float16* __restrict__ Bm,
               const float* __restrict__ bias, void* __restrict__ Cp,
               int M, int N, int K)
{
  (void)M;
  __shared__ _Float16 bstage[2][32 * 64];   // 2 x 4KB
  const int tid  = threadIdx.y * 32 + threadIdx.x;
  const int lane = threadIdx.x;
  const int hh   = lane >> 4;           // lane half selects K sub-chunk
  const int l    = lane & 15;
  const int m0   = blockIdx.y * 128 + threadIdx.y * 32;
  const int n0   = blockIdx.x * 64;
  // LDS byte address of the staging buffer (ISA: LDS addr = flat[31:0])
  const unsigned bbase = (unsigned)(uintptr_t)(&bstage[0][0]);

  // stage one 32(K) x 64(N) B tile into LDS buffer `buf` (256 x 16B chunks,
  // 2 chunks per thread), row-major within the tile
  auto stage = [&](int buf, int k) {
#pragma unroll
    for (int j = 0; j < 2; ++j) {
      const int ch  = tid + j * 128;
      const int row = ch >> 3;
      const int cc  = (ch & 7) * 8;
      async_copy_b128(bbase + (unsigned)(buf * 4096 + ch * 16),
                      Bm + (size_t)(k + row) * N + n0 + cc);
    }
  };

  v8f acc[8];
#pragma unroll
  for (int t = 0; t < 8; ++t)
#pragma unroll
    for (int r = 0; r < 8; ++r) acc[t][r] = 0.f;

  // A 16x32 f16 fragment: lane<16 -> row m0+l, K {k..k+7, k+16..k+23};
  //                       lane>=16 -> same row, K {k+8..k+15, k+24..k+31}
  const _Float16* arow0 = A + (size_t)(m0 + l) * K + hh * 8;
  const _Float16* arow1 = arow0 + (size_t)16 * K;

  stage(0, 0);
  int cur = 0;
  for (int k = 0; k < K; k += 32) {
    wait_async0();        // this wave's copies into bstage[cur] are done
    __syncthreads();      // everyone's copies visible; prev reads of cur^1 done
    if (k + 32 < K) stage(cur ^ 1, k + 32);

    __builtin_prefetch(arow0 + k + 32, 0, 3);
    v8h a0lo = *(const v8h*)(arow0 + k);
    v8h a0hi = *(const v8h*)(arow0 + k + 16);
    v8h a1lo = *(const v8h*)(arow1 + k);
    v8h a1hi = *(const v8h*)(arow1 + k + 16);
    v16h a0 = __builtin_shufflevector(a0lo, a0hi,
        0,1,2,3,4,5,6,7,8,9,10,11,12,13,14,15);
    v16h a1 = __builtin_shufflevector(a1lo, a1hi,
        0,1,2,3,4,5,6,7,8,9,10,11,12,13,14,15);

    // B 32x16 fragments from LDS: lane = N column, 16 halves = K {8hh..8hh+7, 16+8hh..16+8hh+7}
    const _Float16* bb = &bstage[cur][hh * 8 * 64 + l];
    v16h bf[4];
#pragma unroll
    for (int t = 0; t < 4; ++t) {
#pragma unroll
      for (int s = 0; s < 16; ++s) {
        const int kk = (s < 8) ? s : (16 + s - 8);
        bf[t][s] = bb[kk * 64 + t * 16];
      }
    }
#pragma unroll
    for (int t = 0; t < 4; ++t) {
      acc[t]     = __builtin_amdgcn_wmma_f32_16x16x32_f16(
          false, a0, false, bf[t], (short)0, acc[t],     false, false);
      acc[t + 4] = __builtin_amdgcn_wmma_f32_16x16x32_f16(
          false, a1, false, bf[t], (short)0, acc[t + 4], false, false);
    }
    cur ^= 1;
  }

#pragma unroll
  for (int t = 0; t < 4; ++t) {
    const int col = n0 + t * 16 + l;
    const float bv = bias[col];
#pragma unroll
    for (int r = 0; r < 8; ++r) {
      const int row = m0 + (hh ? (r + 8) : r);
      const float v0 = acc[t][r] + bv;
      const float v1 = acc[t + 4][r] + bv;
      if constexpr (OUT_F16) {
        ((_Float16*)Cp)[(size_t)row * N + col]        = (_Float16)v0;
        ((_Float16*)Cp)[(size_t)(row + 16) * N + col] = (_Float16)v1;
      } else {
        ((float*)Cp)[(size_t)row * N + col]        = v0;
        ((float*)Cp)[(size_t)(row + 16) * N + col] = v1;
      }
    }
  }
}

// ---------------- masked softmax over N*NP=64 per (b,q,h) -------------------
__global__ __launch_bounds__(256)
void softmax_kernel(const float* __restrict__ logits,
                    const int* __restrict__ mask,
                    float* __restrict__ attw)
{
  const int t   = blockIdx.x * 8 + threadIdx.y;  // 0..32767 = (b,q,h)
  const int b   = t >> 14;
  const int rem = t & 16383;
  const int q   = rem >> 4;
  const int h   = rem & 15;
  const int lane = threadIdx.x;
  // logits row has MAXN*NP=80 per head; keep first N*NP=64
  const float* lrow = logits + ((size_t)(b * kLq + q)) * kLogC + h * (kMAXN * kNP);
  const int e0 = lane, e1 = lane + 32;
  const float l0 = (mask[b * kN + (e0 >> 3)] > 0) ? lrow[e0] : -1e9f;
  const float l1 = (mask[b * kN + (e1 >> 3)] > 0) ? lrow[e1] : -1e9f;
  float m = fmaxf(l0, l1);
#pragma unroll
  for (int off = 16; off > 0; off >>= 1) m = fmaxf(m, __shfl_xor(m, off, 32));
  const float x0 = __expf(l0 - m), x1 = __expf(l1 - m);
  float s = x0 + x1;
#pragma unroll
  for (int off = 16; off > 0; off >>= 1) s += __shfl_xor(s, off, 32);
  const float inv = 1.f / s;
  float* arow = attw + (((size_t)(b * kLq + q)) * kNH + h) * 64;
  arow[e0] = x0 * inv;
  arow[e1] = x1 * inv;
}

// ---------------- deformable bilinear sampling + weighted sum -> f16 --------
// One wave per (b,q,h); lanes carry head-dim d = lane, lane+32. All sample
// math is lane-uniform (no divergence inside the wave).
__global__ __launch_bounds__(128)
void sample_kernel(const float* __restrict__ value,   // (B,N,256,1024) f32
                   const float* __restrict__ offs,    // (B,Lq,256) f32
                   const float* __restrict__ attw,    // (B,Lq,16,64) f32
                   _Float16* __restrict__ outH)       // (B,Lq,1024) f16
{
  const int t   = blockIdx.x * 4 + threadIdx.y;
  const int b   = t >> 14;
  const int rem = t & 16383;
  const int q   = rem >> 4;
  const int h   = rem & 15;
  const int lane = threadIdx.x;
  const float refx = ((float)(q & 31) + 0.5f) * (1.f / 32.f);
  const float refy = ((float)(q >> 5) + 0.5f) * (1.f / 32.f);
  const float* orow = offs + ((size_t)(b * kLq + q)) * kOffC + h * (kNP * 2);
  const float* arow = attw + (((size_t)(b * kLq + q)) * kNH + h) * 64;
  float acc0 = 0.f, acc1 = 0.f;
  for (int i = 0; i < kN; ++i) {
    const float* vbase = value + (((size_t)b * kN + i) * kTok) * (size_t)kDM + h * kHD;
    for (int p = 0; p < kNP; ++p) {
      const float w = arow[i * kNP + p];
      const float x = (refx + orow[p * 2 + 0] * (1.f / 16.f)) * 16.f - 0.5f;
      const float y = (refy + orow[p * 2 + 1] * (1.f / 16.f)) * 16.f - 0.5f;
      const float xf = floorf(x), yf = floorf(y);
      const float wx = x - xf, wy = y - yf;
      const int x0 = (int)xf, y0 = (int)yf;
      const float cw[4] = {(1.f - wx) * (1.f - wy), wx * (1.f - wy),
                           (1.f - wx) * wy,         wx * wy};
      const int xs[4] = {x0, x0 + 1, x0, x0 + 1};
      const int ys[4] = {y0, y0, y0 + 1, y0 + 1};
#pragma unroll
      for (int cidx = 0; cidx < 4; ++cidx) {
        const int xi = xs[cidx], yi = ys[cidx];
        if (xi >= 0 && xi < kHv && yi >= 0 && yi < kHv) {   // lane-uniform
          const float* vrow = vbase + (size_t)(yi * kHv + xi) * kDM;
          const float ww = w * cw[cidx];
          acc0 += ww * vrow[lane];
          acc1 += ww * vrow[lane + 32];
        }
      }
    }
  }
  _Float16* dst = outH + ((size_t)(b * kLq + q)) * kDM + h * kHD;
  dst[lane]      = (_Float16)acc0;
  dst[lane + 32] = (_Float16)acc1;
}

// ---------------- (B,Lq,C) -> (B,C,H,W) -------------------------------------
__global__ void transpose_out_kernel(const float* __restrict__ y,
                                     float* __restrict__ out)
{
  int i = blockIdx.x * 256 + threadIdx.x;
  if (i < kB * kQD * kLq) {
    int b = i / (kQD * kLq);
    int r = i % (kQD * kLq);
    int c = r / kLq;
    int q = r % kLq;
    out[i] = y[((size_t)(b * kLq + q)) * kQD + c];
  }
}

extern "C" void kernel_launch(void* const* d_in, const int* in_sizes, int n_in,
                              void* d_out, int out_size, void* d_ws, size_t ws_size,
                              hipStream_t stream) {
  (void)in_sizes; (void)n_in; (void)out_size; (void)ws_size;
  const float* sample  = (const float*)d_in[0];
  const float* ms_feat = (const float*)d_in[1];
  const int*   mask    = (const int*)d_in[2];
  // d_in[3] spatial_shapes: compile-time constant (16x16), unused on device
  const float* ln_q_w = (const float*)d_in[4];
  const float* ln_q_b = (const float*)d_in[5];
  const float* ln_f_w = (const float*)d_in[6];
  const float* ln_f_b = (const float*)d_in[7];
  const float* w_off  = (const float*)d_in[8];
  const float* b_off  = (const float*)d_in[9];
  const float* w_attn = (const float*)d_in[10];
  const float* b_attn = (const float*)d_in[11];
  const float* w_val  = (const float*)d_in[12];
  const float* b_val  = (const float*)d_in[13];
  const float* w_out  = (const float*)d_in[14];
  const float* b_out  = (const float*)d_in[15];
  const float* conv_w = (const float*)d_in[16];
  const float* conv_b = (const float*)d_in[17];
  float* out = (float*)d_out;

  char* ws = (char*)d_ws;
  size_t cur = 0;
  auto alloc = [&](size_t bytes) -> void* {
    void* p = ws + cur;
    cur += (bytes + 255) & ~(size_t)255;
    return p;
  };

  const int MRows = kB * kN * kTok;   // 4096 feature rows
  const int QRows = kB * kLq;         // 2048 query rows

  _Float16* qH     = (_Float16*)alloc((size_t)QRows * kQD * 2);
  _Float16* featH  = (_Float16*)alloc((size_t)MRows * kFD * 2);
  _Float16* wvalH  = (_Float16*)alloc((size_t)kFD * kDM * 2);
  _Float16* woffH  = (_Float16*)alloc((size_t)kQD * kOffC * 2);
  _Float16* wattnH = (_Float16*)alloc((size_t)kQD * kLogC * 2);
  _Float16* woutH  = (_Float16*)alloc((size_t)kDM * kQD * 2);
  _Float16* convTH = (_Float16*)alloc((size_t)kQD * kQD * 2);
  float*    value  = (float*)alloc((size_t)MRows * kDM * 4);
  float*    offsb  = (float*)alloc((size_t)QRows * kOffC * 4);
  float*    logits = (float*)alloc((size_t)QRows * kLogC * 4);
  float*    attw   = (float*)alloc((size_t)QRows * kNH * 64 * 4);
  _Float16* outH   = (_Float16*)alloc((size_t)QRows * kDM * 2);
  _Float16* aoutH  = (_Float16*)alloc((size_t)QRows * kQD * 2);
  float*    ybuf   = (float*)alloc((size_t)QRows * kQD * 4);

  prep_query_kernel<<<QRows, kQD, 0, stream>>>(sample, ln_q_w, ln_q_b, qH);
  prep_feat_kernel<<<MRows, 256, 0, stream>>>(ms_feat, ln_f_w, ln_f_b, featH);

  cvt_f16_kernel<<<(kFD * kDM) / 256, 256, 0, stream>>>(w_val, wvalH, kFD * kDM);
  cvt_f16_kernel<<<(kQD * kOffC) / 256, 256, 0, stream>>>(w_off, woffH, kQD * kOffC);
  cvt_f16_kernel<<<(kQD * kLogC) / 256, 256, 0, stream>>>(w_attn, wattnH, kQD * kLogC);
  cvt_f16_kernel<<<(kDM * kQD) / 256, 256, 0, stream>>>(w_out, woutH, kDM * kQD);
  cvt_f16_transpose_kernel<<<(kQD * kQD) / 256, 256, 0, stream>>>(conv_w, convTH, kQD, kQD);

  const dim3 gblk(32, 4);
  // value = LN(feat) @ w_val + b_val            (4096 x 1024 x K=1024)
  gemm_wmma<false><<<dim3(kDM / 64, MRows / 128), gblk, 0, stream>>>(
      featH, wvalH, b_val, value, MRows, kDM, kFD);
  // offs = q @ w_off + b_off                    (2048 x 256 x K=320)
  gemm_wmma<false><<<dim3(kOffC / 64, QRows / 128), gblk, 0, stream>>>(
      qH, woffH, b_off, offsb, QRows, kOffC, kQD);
  // logits = q @ w_attn + b_attn                (2048 x 1280 x K=320)
  gemm_wmma<false><<<dim3(kLogC / 64, QRows / 128), gblk, 0, stream>>>(
      qH, wattnH, b_attn, logits, QRows, kLogC, kQD);

  softmax_kernel<<<(kB * kLq * kNH) / 8, dim3(32, 8), 0, stream>>>(logits, mask, attw);
  sample_kernel<<<(kB * kLq * kNH) / 4, dim3(32, 4), 0, stream>>>(value, offsb, attw, outH);

  // attn_out = out @ w_out + b_out  (f16 result feeds next GEMM directly)
  gemm_wmma<true><<<dim3(kQD / 64, QRows / 128), gblk, 0, stream>>>(
      outH, woutH, b_out, aoutH, QRows, kQD, kDM);
  // y = attn_out @ conv_w^T + conv_b
  gemm_wmma<false><<<dim3(kQD / 64, QRows / 128), gblk, 0, stream>>>(
      aoutH, convTH, conv_b, ybuf, QRows, kQD, kQD);

  transpose_out_kernel<<<(kB * kQD * kLq) / 256, 256, 0, stream>>>(ybuf, out);
}
